// TransformerEncoderLayer_26070451486983
// MI455X (gfx1250) — matrix operand order repaired and checked by reading, not compiled, not run on hardware
//
#include <hip/hip_runtime.h>
#include <cmath>

// ---------------------------------------------------------------------------
// Transformer encoder layer for MI455X (gfx1250, wave32, WMMA).
// All matmuls run through V_WMMA_F32_16X16X32_F16 with f16 operands / f32 acc.
// ---------------------------------------------------------------------------

typedef __attribute__((ext_vector_type(16))) _Float16 v16h;
typedef __attribute__((ext_vector_type(8)))  _Float16 v8h;
typedef __attribute__((ext_vector_type(8)))  float    v8f;

#define MODE_F16_BIAS   0   // outH = (acc + bias) * scale          (q, k, ctx)
#define MODE_GELU_F16   1   // outH = gelu(acc + bias)              (ffn1)
#define MODE_F32_RESID  2   // outF = acc + bias + resid            (wo, ffn2)
#define MODE_F32_SCALE  3   // outF = acc * scale                   (scores)
#define MODE_VT_F16     4   // outH[transposed per-head V layout]   (v proj)

// ---------------------------------------------------------------------------
// Generic batched WMMA GEMM:  C[M,N] = A[M,K] * Bt[N,K]^T (+bias)(+epilogue)
//  - A is f16, or f32 converted on load (AF32, used for softmax probs).
//  - Bt stored transposed (N x K): every B-fragment = two 8-halfword runs (k, k+16)
//    per lane = two global_load_b128, matching the WMMA B lane layout.
//  - Block: 256 threads = 8 waves laid out WM x WN; wave tile 32(M) x 64(N)
//    => 8 v8f accumulators. WM/WN chosen so block tile exactly divides N:
//    no guards, no EXEC masking around WMMA (EXEC must be all-1s).
//  - K is always a multiple of 64 here: K-loop unrolled by 2 with register
//    double-buffering so fragment loads overlap the previous step's WMMAs.
// ---------------------------------------------------------------------------
template<int WM, int WN, bool AF32, int MODE>
__global__ __launch_bounds__(256)
void wmma_gemm_kernel(const _Float16* __restrict__ Ah,
                      const float*    __restrict__ Af,
                      const _Float16* __restrict__ Bt,
                      const float*    __restrict__ bias,
                      const float*    __restrict__ resid,
                      _Float16*       __restrict__ outH,
                      float*          __restrict__ outF,
                      int K, int lda, int ldb, int ldc,
                      long long sAb, long long sAh,
                      long long sBb, long long sBh,
                      long long sCb, long long sCh,
                      int nh, float scale)
{
    const int z = blockIdx.z;
    const int b = z / nh;
    const int h = z % nh;
    const long long offA = (long long)b * sAb + (long long)h * sAh;
    const long long offB = (long long)b * sBb + (long long)h * sBh;
    const long long offC = (long long)b * sCb + (long long)h * sCh;

    const int lane  = threadIdx.x & 31;
    const int wave  = threadIdx.x >> 5;
    const int waveM = wave / WN;
    const int waveN = wave % WN;
    const int blockRow = blockIdx.y * (WM * 32) + waveM * 32;
    const int blockCol = blockIdx.x * (WN * 64) + waveN * 64;

    const int lm  = lane & 15;
    const int aKb = (lane < 16) ? 0 : 8;     // A frag K sub-base per lane half
    const int bKb = (lane < 16) ? 0 : 8;     // B frag K sub-base per lane half

    // Per-lane fragment base pointers (k advances via immediate offsets).
    const _Float16* aBaseH[2];
    const float*    aBaseF[2];
#pragma unroll
    for (int mi = 0; mi < 2; ++mi) {
        const long long r = blockRow + mi * 16 + lm;
        if (AF32) aBaseF[mi] = Af + offA + r * (long long)lda + aKb;
        else      aBaseH[mi] = Ah + offA + r * (long long)lda + aKb;
    }
    const _Float16* bBase[4];
#pragma unroll
    for (int ni = 0; ni < 4; ++ni) {
        const long long n = blockCol + ni * 16 + lm;
        bBase[ni] = Bt + offB + n * (long long)ldb + bKb;
    }

    v8f acc[2][4];
#pragma unroll
    for (int mi = 0; mi < 2; ++mi)
#pragma unroll
        for (int ni = 0; ni < 4; ++ni)
            acc[mi][ni] = (v8f){0.f,0.f,0.f,0.f,0.f,0.f,0.f,0.f};

    // ---- fragment loaders ----
    auto loadA = [&](v16h (&af)[2], int k0) {
#pragma unroll
        for (int mi = 0; mi < 2; ++mi) {
            if (AF32) {
                const float* p = aBaseF[mi] + k0;
#pragma unroll
                for (int i = 0; i < 8; ++i) {
                    af[mi][i]     = (_Float16)p[i];
                    af[mi][8 + i] = (_Float16)p[16 + i];
                }
            } else {
                const _Float16* p = aBaseH[mi] + k0;
                v8h c0 = *reinterpret_cast<const v8h*>(p);
                v8h c1 = *reinterpret_cast<const v8h*>(p + 16);
#pragma unroll
                for (int i = 0; i < 8; ++i) { af[mi][i] = c0[i]; af[mi][8 + i] = c1[i]; }
            }
        }
    };
    auto loadB = [&](v16h (&bf)[4], int k0) {
#pragma unroll
        for (int ni = 0; ni < 4; ++ni) {
            const _Float16* p = bBase[ni] + k0;
            v8h c0 = *reinterpret_cast<const v8h*>(p);
            v8h c1 = *reinterpret_cast<const v8h*>(p + 16);
#pragma unroll
            for (int i = 0; i < 8; ++i) { bf[ni][i] = c0[i]; bf[ni][8 + i] = c1[i]; }
        }
    };
    auto mma = [&](const v16h (&af)[2], const v16h (&bf)[4]) {
#pragma unroll
        for (int ni = 0; ni < 4; ++ni) {
            acc[0][ni] = __builtin_amdgcn_wmma_f32_16x16x32_f16(
                false, af[0], false, bf[ni], (short)0, acc[0][ni], false, false);
            acc[1][ni] = __builtin_amdgcn_wmma_f32_16x16x32_f16(
                false, af[1], false, bf[ni], (short)0, acc[1][ni], false, false);
        }
    };

    // ---- main loop: K % 64 == 0; double-buffered, loads overlap WMMAs ----
    v16h a0[2], a1[2], b0[4], b1[4];
    loadA(a0, 0);
    loadB(b0, 0);
    for (int k0 = 0; k0 < K; k0 += 64) {
        loadA(a1, k0 + 32);            // stage next half-step
        loadB(b1, k0 + 32);
        mma(a0, b0);
        if (k0 + 64 < K) {             // scalar-uniform branch: EXEC untouched
            loadA(a0, k0 + 64);
            loadB(b0, k0 + 64);
        }
        mma(a1, b1);
    }

    // ---- epilogue ----
    const int rowHi = (lane >= 16) ? 8 : 0;
#pragma unroll
    for (int mi = 0; mi < 2; ++mi) {
#pragma unroll
        for (int ni = 0; ni < 4; ++ni) {
            const int c = blockCol + ni * 16 + lm;
#pragma unroll
            for (int v = 0; v < 8; ++v) {
                const long long r = blockRow + mi * 16 + rowHi + v;
                float val = acc[mi][ni][v];
                if (bias) val += bias[c];
                const long long cidx = offC + r * (long long)ldc + c;
                if (MODE == MODE_F16_BIAS) {
                    outH[cidx] = (_Float16)(val * scale);
                } else if (MODE == MODE_GELU_F16) {
                    float g = 0.5f * val * (1.0f + erff(val * 0.70710678118f));
                    outH[cidx] = (_Float16)g;
                } else if (MODE == MODE_F32_RESID) {
                    outF[cidx] = val + resid[cidx];
                } else if (MODE == MODE_F32_SCALE) {
                    outF[cidx] = val * scale;
                } else { // MODE_VT_F16: write V transposed per head: (b,h,d,n)
                    const long long bb = r >> 10;        // token row -> batch
                    const long long n  = r & 1023;       // token index
                    const int h2 = c >> 6;               // head
                    const int d2 = c & 63;               // dim within head
                    outH[((bb * nh + h2) * 64 + d2) * 1024 + n] = (_Float16)val;
                }
            }
        }
    }
}

// ---------------------------------------------------------------------------
// Row LayerNorm (H=768) -> f16 output for the WMMA GEMMs. One block per row.
// ---------------------------------------------------------------------------
__global__ __launch_bounds__(256)
void layernorm_f16_kernel(const float* __restrict__ x,
                          const float* __restrict__ g,
                          const float* __restrict__ be,
                          _Float16* __restrict__ out, int H)
{
    __shared__ float red[256];
    const long long row = blockIdx.x;
    const float* xr = x + row * (long long)H;
    const int tid = threadIdx.x;

    float s = 0.f;
    for (int i = tid; i < H; i += 256) s += xr[i];
    red[tid] = s;
    for (int st = 128; st > 0; st >>= 1) {
        __syncthreads();
        if (tid < st) red[tid] += red[tid + st];
    }
    __syncthreads();
    const float mu = red[0] / (float)H;
    __syncthreads();

    float v = 0.f;
    for (int i = tid; i < H; i += 256) { float d = xr[i] - mu; v += d * d; }
    red[tid] = v;
    for (int st = 128; st > 0; st >>= 1) {
        __syncthreads();
        if (tid < st) red[tid] += red[tid + st];
    }
    __syncthreads();
    const float rstd = rsqrtf(red[0] / (float)H + 1e-6f);

    _Float16* orow = out + row * (long long)H;
    for (int i = tid; i < H; i += 256)
        orow[i] = (_Float16)((xr[i] - mu) * rstd * g[i] + be[i]);
}

// ---------------------------------------------------------------------------
// Row softmax over the 1024-wide score rows (in place, fp32 in d_out).
// ---------------------------------------------------------------------------
__global__ __launch_bounds__(256)
void softmax_kernel(float* __restrict__ w, int Ncol)
{
    __shared__ float red[256];
    const long long row = blockIdx.x;
    float* p = w + row * (long long)Ncol;
    const int tid = threadIdx.x;

    float m = -3.4e38f;
    for (int i = tid; i < Ncol; i += 256) m = fmaxf(m, p[i]);
    red[tid] = m;
    for (int st = 128; st > 0; st >>= 1) {
        __syncthreads();
        if (tid < st) red[tid] = fmaxf(red[tid], red[tid + st]);
    }
    __syncthreads();
    const float mx = red[0];
    __syncthreads();

    float s = 0.f;
    for (int i = tid; i < Ncol; i += 256) {
        float e = expf(p[i] - mx);
        p[i] = e;
        s += e;
    }
    red[tid] = s;
    for (int st = 128; st > 0; st >>= 1) {
        __syncthreads();
        if (tid < st) red[tid] += red[tid + st];
    }
    __syncthreads();
    const float inv = 1.0f / red[0];
    for (int i = tid; i < Ncol; i += 256) p[i] *= inv;
}

// ---------------------------------------------------------------------------
// Weight convert + transpose: W (K x N, f32, row major) -> Wt (N x K, f16).
// Gives every WMMA B-fragment two contiguous 8-halfword runs per lane.
// ---------------------------------------------------------------------------
__global__ __launch_bounds__(256)
void transpose_cvt_kernel(const float* __restrict__ W,
                          _Float16* __restrict__ Wt, int K, int N)
{
    __shared__ _Float16 s[32][33];
    const int tx = threadIdx.x & 31;
    const int ty = threadIdx.x >> 5;       // 0..7
    const int n0 = blockIdx.x * 32;
    const int k0 = blockIdx.y * 32;
#pragma unroll
    for (int j = 0; j < 32; j += 8)
        s[ty + j][tx] = (_Float16)W[(long long)(k0 + ty + j) * N + n0 + tx];
    __syncthreads();
#pragma unroll
    for (int j = 0; j < 32; j += 8)
        Wt[(long long)(n0 + ty + j) * K + k0 + tx] = s[tx][ty + j];
}

// ---------------------------------------------------------------------------
// Host side
// ---------------------------------------------------------------------------
template<int WM, int WN, bool AF32, int MODE>
static void launch_gemm(hipStream_t stream,
                        const _Float16* Ah, const float* Af, const _Float16* Bt,
                        const float* bias, const float* resid,
                        _Float16* outH, float* outF,
                        int M, int N, int K, int lda, int ldb, int ldc,
                        long long sAb, long long sAh, long long sBb, long long sBh,
                        long long sCb, long long sCh,
                        int batch, int nh, float scale)
{
    dim3 grid(N / (WN * 64), M / (WM * 32), batch);
    wmma_gemm_kernel<WM, WN, AF32, MODE><<<grid, dim3(256), 0, stream>>>(
        Ah, Af, Bt, bias, resid, outH, outF,
        K, lda, ldb, ldc, sAb, sAh, sBb, sBh, sCb, sCh, nh, scale);
}

extern "C" void kernel_launch(void* const* d_in, const int* in_sizes, int n_in,
                              void* d_out, int out_size, void* d_ws, size_t ws_size,
                              hipStream_t stream)
{
    const int B = 8, NT = 1024, H = 768, NH = 12, FF = 3072;
    const int M = B * NT;                       // 8192 token rows
    (void)in_sizes; (void)n_in; (void)out_size; (void)ws_size;

    const float* x   = (const float*)d_in[0];
    const float* wq  = (const float*)d_in[1];
    const float* bq  = (const float*)d_in[2];
    const float* wk  = (const float*)d_in[3];
    const float* bk  = (const float*)d_in[4];
    const float* wv  = (const float*)d_in[5];
    const float* bv  = (const float*)d_in[6];
    const float* wo  = (const float*)d_in[7];
    const float* bo  = (const float*)d_in[8];
    const float* w1  = (const float*)d_in[9];
    const float* b1  = (const float*)d_in[10];
    const float* w2  = (const float*)d_in[11];
    const float* b2  = (const float*)d_in[12];
    const float* g1  = (const float*)d_in[13];
    const float* be1 = (const float*)d_in[14];
    const float* g2  = (const float*)d_in[15];
    const float* be2 = (const float*)d_in[16];

    float* xout = (float*)d_out;                          // (8,1024,768)
    float* wout = xout + (size_t)M * H;                   // (8,12,1024,1024)

    // ---- carve workspace (aligned to 256B) ----
    char* base = (char*)d_ws;
    size_t off = 0;
    auto carve = [&](size_t bytes) -> void* {
        void* p = base + off;
        off = (off + bytes + 255) & ~(size_t)255;
        return p;
    };
    _Float16* x1h  = (_Float16*)carve((size_t)M * H * 2);
    _Float16* wqt  = (_Float16*)carve((size_t)H * H * 2);
    _Float16* wkt  = (_Float16*)carve((size_t)H * H * 2);
    _Float16* wvt  = (_Float16*)carve((size_t)H * H * 2);
    _Float16* wot  = (_Float16*)carve((size_t)H * H * 2);
    _Float16* w1t  = (_Float16*)carve((size_t)H * FF * 2);
    _Float16* w2t  = (_Float16*)carve((size_t)FF * H * 2);
    _Float16* qh   = (_Float16*)carve((size_t)M * H * 2);
    _Float16* kh   = (_Float16*)carve((size_t)M * H * 2);
    _Float16* vth  = (_Float16*)carve((size_t)M * H * 2);   // (B,NH,64,1024)
    _Float16* ctxh = (_Float16*)carve((size_t)M * H * 2);
    float*    xmid = (float*)   carve((size_t)M * H * 4);
    _Float16* x2h  = (_Float16*)carve((size_t)M * H * 2);
    _Float16* hh   = (_Float16*)carve((size_t)M * FF * 2);

    // ---- 1. weight transpose/convert to f16 (Wt is N x K) ----
    transpose_cvt_kernel<<<dim3(H/32,  H/32),  256, 0, stream>>>(wq, wqt, H, H);
    transpose_cvt_kernel<<<dim3(H/32,  H/32),  256, 0, stream>>>(wk, wkt, H, H);
    transpose_cvt_kernel<<<dim3(H/32,  H/32),  256, 0, stream>>>(wv, wvt, H, H);
    transpose_cvt_kernel<<<dim3(H/32,  H/32),  256, 0, stream>>>(wo, wot, H, H);
    transpose_cvt_kernel<<<dim3(FF/32, H/32),  256, 0, stream>>>(w1, w1t, H, FF);
    transpose_cvt_kernel<<<dim3(H/32,  FF/32), 256, 0, stream>>>(w2, w2t, FF, H);

    // ---- 2. LN1 ----
    layernorm_f16_kernel<<<M, 256, 0, stream>>>(x, g1, be1, x1h, H);

    // ---- 3. Q/K/V projections (V written transposed per head) ----
    launch_gemm<4,2,false,MODE_F16_BIAS>(stream, x1h, nullptr, wqt, bq, nullptr, qh, nullptr,
                M, H, H, H, H, H, 0,0,0,0,0,0, 1, NH, 1.0f);
    launch_gemm<4,2,false,MODE_F16_BIAS>(stream, x1h, nullptr, wkt, bk, nullptr, kh, nullptr,
                M, H, H, H, H, H, 0,0,0,0,0,0, 1, NH, 1.0f);
    launch_gemm<4,2,false,MODE_VT_F16>(stream, x1h, nullptr, wvt, bv, nullptr, vth, nullptr,
                M, H, H, H, H, H, 0,0,0,0,0,0, 1, NH, 1.0f);

    // ---- 4. scores = Q K^T / sqrt(H)  (K rows are already "Bt" layout) ----
    const float sscale = 1.0f / sqrtf((float)H);
    launch_gemm<4,2,false,MODE_F32_SCALE>(stream, qh, nullptr, kh, nullptr, nullptr, nullptr, wout,
                NT, NT, 64, H, H, NT,
                (long long)NT * H, 64,               // A: per-b, per-h offsets
                (long long)NT * H, 64,               // B: same layout as A
                (long long)NH * NT * NT, (long long)NT * NT,  // C: (b,h,1024,1024)
                B * NH, NH, sscale);

    // ---- 5. softmax over keys (in place in d_out) ----
    softmax_kernel<<<B * NH * NT, 256, 0, stream>>>(wout, NT);

    // ---- 6. ctx = P @ V   (P f32 converted on load; V pre-transposed)
    //      N=64 => 8x1 wave layout, 256x64 block tile, no guards needed ----
    launch_gemm<8,1,true,MODE_F16_BIAS>(stream, nullptr, wout, vth, nullptr, nullptr, ctxh, nullptr,
                NT, 64, NT, NT, NT, H,
                (long long)NH * NT * NT, (long long)NT * NT,  // A over (b,h)
                (long long)NH * 64 * NT, (long long)64 * NT,  // B over (b,h)
                (long long)NT * H, 64,                        // C back to (token, head*64)
                B * NH, NH, 1.0f);

    // ---- 7. output projection + residual ----
    launch_gemm<4,2,false,MODE_F32_RESID>(stream, ctxh, nullptr, wot, bo, x, nullptr, xmid,
                M, H, H, H, H, H, 0,0,0,0,0,0, 1, NH, 1.0f);

    // ---- 8. LN2 ----
    layernorm_f16_kernel<<<M, 256, 0, stream>>>(xmid, g2, be2, x2h, H);

    // ---- 9. FFN: gelu(x2 @ w1 + b1) @ w2 + b2 + residual ----
    launch_gemm<4,2,false,MODE_GELU_F16>(stream, x2h, nullptr, w1t, b1, nullptr, hh, nullptr,
                M, FF, H, H, H, FF, 0,0,0,0,0,0, 1, NH, 1.0f);
    launch_gemm<4,2,false,MODE_F32_RESID>(stream, hh, nullptr, w2t, b2, xmid, nullptr, xout,
                M, H, FF, FF, FF, H, 0,0,0,0,0,0, 1, NH, 1.0f);
}
